// PositionAwareAttention_42545946034878
// MI455X (gfx1250) — matrix operand order repaired and checked by reading, not compiled, not run on hardware
//
#include <hip/hip_runtime.h>
#include <hip/hip_bf16.h>

typedef __attribute__((ext_vector_type(16))) __bf16 v16bf;
typedef __attribute__((ext_vector_type(8)))  float  v8f;

__device__ __forceinline__ unsigned short f2bf(float f) {
  unsigned int u = __builtin_bit_cast(unsigned int, f);
  u += 0x7FFFu + ((u >> 16) & 1u);          // round-to-nearest-even
  return (unsigned short)(u >> 16);
}

union FragBF { uint4 q[2]; v16bf v; };

// A fragment (16x32 bf16): lane lr = row m; elements 0..7 -> k = 8*hi..8*hi+7,
// elements 8..15 -> k = 16+8*hi..16+8*hi+7 (ISA 16-bit A-matrix layout).
__device__ __forceinline__ v16bf load_a_frag(const unsigned short* base, int lr, int hi) {
  FragBF u;
  const unsigned short* row = base + lr * 40;
  u.q[0] = *(const uint4*)(row + 8 * hi);
  u.q[1] = *(const uint4*)(row + 16 + 8 * hi);
  return u.v;
}
// B fragment (32x16 bf16): lane lr = col n; elements 0..7 -> k = 8*hi..8*hi+7,
// elements 8..15 -> k = 16+8*hi..16+8*hi+7.
// LDS tile stored as [n][k].
__device__ __forceinline__ v16bf load_b_frag(const unsigned short* base, int lr, int hi) {
  FragBF u;
  const unsigned short* row = base + lr * 40;
  u.q[0] = *(const uint4*)(row + 8 * hi);
  u.q[1] = *(const uint4*)(row + 16 + 8 * hi);
  return u.v;
}

// CDNA5 async copy: global -> LDS, 16B per lane, tracked by ASYNCcnt.
// ldsByteOff is the wave-relative LDS byte offset (low 32 bits of generic ptr).
__device__ __forceinline__ void async_b128(unsigned ldsByteOff, const void* g) {
  asm volatile("global_load_async_to_lds_b128 %0, %1, off"
               :: "v"(ldsByteOff), "v"((unsigned long long)(size_t)g)
               : "memory");
}
__device__ __forceinline__ void wait_async0() {
  asm volatile("s_wait_asynccnt 0x0" ::: "memory");
}

// ---------------------------------------------------------------------------
// 128x128-tile GEMM:  C[b] = scale * (A[b] @ B[b] + bias)
//   A: [M][K] bf16(ushort) staged via global_load_async_to_lds_b128,
//      or f32 staged via registers (loaded one iteration ahead).
//   B: bf16 [N][K] row-major (pre-transposed weights / K rows / Vt).
//   8 waves; wave w -> 32-row strip (w>>1), 64-col strip (w&1):
//   2 A-frags x 4 B-frags = 8 WMMAs per 32-deep K-step.
//   Double-buffered LDS, one barrier + one s_wait_asynccnt per K-step.
// ---------------------------------------------------------------------------
template <typename AT, typename OT, bool POSW, bool TRANSC>
__global__ __launch_bounds__(256) void gemm128(
    const AT* __restrict__ A, long long aBatch, int lda,
    const unsigned short* __restrict__ Bm, long long bBatch, int ldb,
    OT* __restrict__ C, long long cBatch, int ldc,
    const float* __restrict__ bias, float scale, int Ksz) {
  constexpr bool A_F32 = (sizeof(AT) == 4);
  constexpr int TILE = 128 * 40;                       // ushorts per tile
  __shared__ __align__(16) unsigned short smem[4 * TILE];  // [buf][A|B]
  const unsigned smemOff = (unsigned)(size_t)smem;     // LDS byte offset of smem

  const int tid  = threadIdx.x;
  const int wave = tid >> 5, lane = tid & 31;
  const int lr = lane & 15, hi = lane >> 4;
  const int mq = wave >> 1;            // 0..3 : 32-row strip
  const int nq = wave & 1;             // 0..1 : 64-col strip
  const int m0 = blockIdx.y * 128, n0 = blockIdx.x * 128;

  A  += (size_t)blockIdx.z * aBatch;
  Bm += (size_t)blockIdx.z * bBatch;
  if constexpr (!TRANSC) C += (size_t)blockIdx.z * cBatch;

  const int srow = tid >> 1;           // 0..127 (staging row)
  const int skc  = (tid & 1) * 16;     // 0 / 16 (staging k chunk)

  const unsigned short* gB = Bm + (size_t)(n0 + srow) * ldb + skc;
  const AT*             gA = A  + (size_t)(m0 + srow) * lda + skc;

  v8f acc[8] = {};
  float regs[16];                      // f32 A staging (A_F32 only)

  auto issueB = [&](int buf, int k0) {
    const unsigned dst = smemOff + (unsigned)(buf * 2 * TILE + TILE + srow * 40 + skc) * 2;
    async_b128(dst,      gB + k0);
    async_b128(dst + 16, gB + k0 + 8);
  };
  auto issueA_bf = [&](int buf, int k0) {
    const unsigned dst = smemOff + (unsigned)(buf * 2 * TILE + srow * 40 + skc) * 2;
    async_b128(dst,      (const unsigned short*)gA + k0);
    async_b128(dst + 16, (const unsigned short*)gA + k0 + 8);
  };
  auto loadA_f32 = [&](int k0) {
    const float* p = (const float*)gA + k0;
#pragma unroll
    for (int i = 0; i < 4; ++i) {
      float4 f = *(const float4*)(p + i * 4);
      regs[i * 4 + 0] = f.x; regs[i * 4 + 1] = f.y;
      regs[i * 4 + 2] = f.z; regs[i * 4 + 3] = f.w;
    }
  };
  auto storeA_f32 = [&](int buf) {
#pragma unroll
    for (int h = 0; h < 2; ++h) {
      union { unsigned short s[8]; uint4 q; } w;
#pragma unroll
      for (int i = 0; i < 8; ++i) w.s[i] = f2bf(regs[h * 8 + i]);
      *(uint4*)&smem[buf * 2 * TILE + srow * 40 + skc + h * 8] = w.q;
    }
  };

  // Prologue: tile 0 -> buffer 0
  if constexpr (A_F32) loadA_f32(0); else issueA_bf(0, 0);
  issueB(0, 0);

  const int nk = Ksz >> 5;
  for (int it = 0; it < nk; ++it) {
    const int buf = it & 1;
    if constexpr (A_F32) storeA_f32(buf);       // regs from previous iteration
    wait_async0();                               // own async writes for tile 'it' done
    __syncthreads();                             // tile 'it' visible; buf^1 free
    if (it + 1 < nk) {
      if constexpr (A_F32) loadA_f32((it + 1) * 32);
      else issueA_bf(buf ^ 1, (it + 1) * 32);
      issueB(buf ^ 1, (it + 1) * 32);
    }
    const unsigned short* aT = smem + buf * 2 * TILE + mq * 32 * 40;
    const unsigned short* bT = smem + buf * 2 * TILE + TILE + nq * 64 * 40;
    v16bf a0 = load_a_frag(aT, lr, hi);
    v16bf a1 = load_a_frag(aT + 16 * 40, lr, hi);
    v16bf b0 = load_b_frag(bT, lr, hi);
    v16bf b1 = load_b_frag(bT + 16 * 40, lr, hi);
    v16bf b2 = load_b_frag(bT + 32 * 40, lr, hi);
    v16bf b3 = load_b_frag(bT + 48 * 40, lr, hi);
    acc[0] = __builtin_amdgcn_wmma_f32_16x16x32_bf16(false, a0, false, b0, (short)0, acc[0], false, false);
    acc[1] = __builtin_amdgcn_wmma_f32_16x16x32_bf16(false, a0, false, b1, (short)0, acc[1], false, false);
    acc[2] = __builtin_amdgcn_wmma_f32_16x16x32_bf16(false, a0, false, b2, (short)0, acc[2], false, false);
    acc[3] = __builtin_amdgcn_wmma_f32_16x16x32_bf16(false, a0, false, b3, (short)0, acc[3], false, false);
    acc[4] = __builtin_amdgcn_wmma_f32_16x16x32_bf16(false, a1, false, b0, (short)0, acc[4], false, false);
    acc[5] = __builtin_amdgcn_wmma_f32_16x16x32_bf16(false, a1, false, b1, (short)0, acc[5], false, false);
    acc[6] = __builtin_amdgcn_wmma_f32_16x16x32_bf16(false, a1, false, b2, (short)0, acc[6], false, false);
    acc[7] = __builtin_amdgcn_wmma_f32_16x16x32_bf16(false, a1, false, b3, (short)0, acc[7], false, false);
  }

  // Epilogue. C-frag: lane lr = col n; VGPR r -> row m = r + 8*hi.
#pragma unroll
  for (int i = 0; i < 2; ++i) {
#pragma unroll
    for (int j = 0; j < 4; ++j) {
      const v8f cv = acc[i * 4 + j];
      const int nG = n0 + nq * 64 + j * 16 + lr;
      const float bval = bias ? bias[nG] : 0.0f;
      const int mBase = m0 + mq * 32 + i * 16 + 8 * hi;
      if constexpr (TRANSC) {            // Vt[b][n][s], b = m>>11, s = m&2047
        union { unsigned short s[8]; uint4 q; } pk;
#pragma unroll
        for (int r = 0; r < 8; ++r) pk.s[r] = f2bf((cv[r] + bval) * scale);
        unsigned short* cb = (unsigned short*)C + (size_t)(mBase >> 11) * cBatch;
        *(uint4*)(cb + (size_t)nG * ldc + (mBase & 2047)) = pk.q;
      } else {
#pragma unroll
        for (int r = 0; r < 8; ++r) {
          const int m = mBase + r;
          float v = (cv[r] + bval) * scale;
          if constexpr (POSW) {
            // pw = (1 + 1.5*exp(-2*|s-t|/2048)) / 2.5
            const float d = fabsf((float)(m - nG)) * (1.0f / 2048.0f);
            v *= 0.4f * (1.0f + 1.5f * __expf(-2.0f * d));
          }
          if constexpr (sizeof(OT) == 2)
            ((unsigned short*)C)[(size_t)m * ldc + nG] = f2bf(v);
          else
            ((float*)C)[(size_t)m * ldc + nG] = v;
        }
      }
    }
  }
}

// xp = bf16(x + pos_emb), pos broadcast over batch (S*D = 2^21)
__global__ __launch_bounds__(256) void addpos(const float* __restrict__ x,
                                              const float* __restrict__ pos,
                                              unsigned short* __restrict__ xp) {
  const size_t i = (size_t)blockIdx.x * 256 + threadIdx.x;
  xp[i] = f2bf(x[i] + pos[i & 2097151u]);
}

// Wt[n][k] = bf16(W[k][n]), D=1024, 32x32 LDS-tiled
__global__ __launch_bounds__(256) void transposeW(const float* __restrict__ W,
                                                  unsigned short* __restrict__ Wt) {
  __shared__ unsigned short tile[32][33];
  const int tx = threadIdx.x & 31, ty = threadIdx.x >> 5;
  const int kb = blockIdx.y * 32, nb = blockIdx.x * 32;
#pragma unroll
  for (int i = 0; i < 4; ++i)
    tile[ty + i * 8][tx] = f2bf(W[(size_t)(kb + ty + i * 8) * 1024 + nb + tx]);
  __syncthreads();
#pragma unroll
  for (int i = 0; i < 4; ++i)
    Wt[(size_t)(nb + ty + i * 8) * 1024 + kb + tx] = tile[tx][ty + i * 8];
}

// In-place row softmax over 2048 columns; one block per (b,s) row.
__global__ __launch_bounds__(256) void softmax_rows(float* __restrict__ attn) {
  float* p = attn + (size_t)blockIdx.x * 2048;
  const int tid = threadIdx.x;
  __shared__ float red[256];
  float v[8];
  float mx = -3.4e38f;
#pragma unroll
  for (int i = 0; i < 8; ++i) { v[i] = p[i * 256 + tid]; mx = fmaxf(mx, v[i]); }
  red[tid] = mx; __syncthreads();
  for (int s = 128; s > 0; s >>= 1) {
    if (tid < s) red[tid] = fmaxf(red[tid], red[tid + s]);
    __syncthreads();
  }
  mx = red[0]; __syncthreads();
  float sum = 0.0f;
#pragma unroll
  for (int i = 0; i < 8; ++i) { v[i] = __expf(v[i] - mx); sum += v[i]; }
  red[tid] = sum; __syncthreads();
  for (int s = 128; s > 0; s >>= 1) {
    if (tid < s) red[tid] += red[tid + s];
    __syncthreads();
  }
  const float inv = 1.0f / red[0];
#pragma unroll
  for (int i = 0; i < 8; ++i) p[i * 256 + tid] = v[i] * inv;
}

extern "C" void kernel_launch(void* const* d_in, const int* in_sizes, int n_in,
                              void* d_out, int out_size, void* d_ws, size_t ws_size,
                              hipStream_t stream) {
  const int S = 2048, D = 1024;
  const size_t M = 8ull * S;                       // B*S = 16384 tokens

  const float* x   = (const float*)d_in[0];
  const float* pos = (const float*)d_in[1];
  const float* Wq  = (const float*)d_in[2];
  const float* bq  = (const float*)d_in[3];
  const float* Wk  = (const float*)d_in[4];
  const float* bk  = (const float*)d_in[5];
  const float* Wv  = (const float*)d_in[6];
  const float* bv  = (const float*)d_in[7];
  const float* Wo  = (const float*)d_in[8];
  const float* bo  = (const float*)d_in[9];

  float* outp = (float*)d_out;                     // [B,S,D] f32
  float* attn = outp + M * D;                      // [B,S,S] f32

  size_t off = 0;
  char* base = (char*)d_ws;
  auto alloc = [&](size_t bytes) -> void* {
    void* p = base + off;
    off += (bytes + 255) & ~(size_t)255;
    return p;
  };
  unsigned short* xp  = (unsigned short*)alloc(M * D * 2);   // bf16 x+pos
  unsigned short* Wtq = (unsigned short*)alloc((size_t)D * D * 2);
  unsigned short* Wtk = (unsigned short*)alloc((size_t)D * D * 2);
  unsigned short* Wtv = (unsigned short*)alloc((size_t)D * D * 2);
  unsigned short* Wto = (unsigned short*)alloc((size_t)D * D * 2);
  unsigned short* Qb  = (unsigned short*)alloc(M * D * 2);   // Q * (1/sqrt(D))
  unsigned short* Kb  = (unsigned short*)alloc(M * D * 2);
  unsigned short* Vt  = (unsigned short*)alloc(M * D * 2);   // Vt[b][d][t]
  unsigned short* mid = (unsigned short*)alloc(M * D * 2);   // attn@V, bf16

  // 1) weight transpose+convert to bf16 [N][K]
  transposeW<<<dim3(32, 32, 1), 256, 0, stream>>>(Wq, Wtq);
  transposeW<<<dim3(32, 32, 1), 256, 0, stream>>>(Wk, Wtk);
  transposeW<<<dim3(32, 32, 1), 256, 0, stream>>>(Wv, Wtv);
  transposeW<<<dim3(32, 32, 1), 256, 0, stream>>>(Wo, Wto);

  // 2) xp = bf16(x + pos)
  addpos<<<dim3((unsigned)(M * D / 256)), 256, 0, stream>>>(x, pos, xp);

  // 3) Q,K,V projections (Q pre-scaled by 1/32; V stored transposed per batch)
  gemm128<unsigned short, unsigned short, false, false>
      <<<dim3(8, 128, 1), 256, 0, stream>>>(xp, 0, D, Wtq, 0, D, Qb, 0, D, bq, 1.0f / 32.0f, D);
  gemm128<unsigned short, unsigned short, false, false>
      <<<dim3(8, 128, 1), 256, 0, stream>>>(xp, 0, D, Wtk, 0, D, Kb, 0, D, bk, 1.0f, D);
  gemm128<unsigned short, unsigned short, false, true>
      <<<dim3(8, 128, 1), 256, 0, stream>>>(xp, 0, D, Wtv, 0, D, Vt, (long long)D * S, S, bv, 1.0f, D);

  // 4) scores = (Q@K^T) * pw  -> attn region (raw, pre-softmax)
  gemm128<unsigned short, float, true, false>
      <<<dim3(16, 16, 8), 256, 0, stream>>>(Qb, (long long)S * D, D, Kb, (long long)S * D, D,
                                            attn, (long long)S * S, S, nullptr, 1.0f, D);

  // 5) softmax in place over attn rows
  softmax_rows<<<dim3(16384), 256, 0, stream>>>(attn);

  // 6) mid = attn @ V   (A = f32 attn via register staging; B = Vt[b][d][t])
  gemm128<float, unsigned short, false, false>
      <<<dim3(8, 16, 8), 256, 0, stream>>>(attn, (long long)S * S, S, Vt, (long long)D * S, S,
                                           mid, (long long)S * D, D, nullptr, 1.0f, S);

  // 7) out = mid @ Wo + bo  -> f32 d_out
  gemm128<unsigned short, float, false, false>
      <<<dim3(8, 128, 1), 256, 0, stream>>>(mid, 0, D, Wto, 0, D, outp, 0, D, bo, 1.0f, D);
}
